// MHA_38817914421604
// MI455X (gfx1250) — compile-verified, matrix operand order checked by reading, not run on hardware
//
#include <hip/hip_runtime.h>
#include <hip/hip_bf16.h>

// ---------------------------------------------------------------------------
// MHA with ALiBi + causal mask for MI455X (gfx1250, wave32, WMMA bf16)
//
//   D_MODEL=2048, N_HEADS=16, HEAD_DIM=128, BATCH=2, SEQ=2048
//
// Pipeline:
//   1) f32 -> bf16 conversion of x and W_q/W_k/W_v/W_o
//   2) QKV projection GEMMs via v_wmma_f32_16x16x32_bf16, 32x64 wave tiles
//      (Q scaled by 1/sqrt(128); V stored transposed [B,H,Dh,S])
//   3) flash-attention (online softmax, 16-row query tiles, 32-key tiles,
//      alibi prefetch via global_prefetch_b8)
//   4) output projection GEMM -> f32 d_out
// ---------------------------------------------------------------------------

typedef __bf16 bf16;
typedef __attribute__((ext_vector_type(16))) __bf16 v16bf;
typedef __attribute__((ext_vector_type(8)))  float  v8f;

#define D_MODEL 2048
#define N_HEADS 16
#define HEAD_DIM 128
#define BATCH 2
#define SEQ 2048
#define ROWS (BATCH * SEQ)       // 4096

static __device__ __forceinline__ float rowmax16(float v) {
  // reduce across a 16-lane half (offsets < 16 stay within the half)
  v = fmaxf(v, __shfl_xor(v, 1, 32));
  v = fmaxf(v, __shfl_xor(v, 2, 32));
  v = fmaxf(v, __shfl_xor(v, 4, 32));
  v = fmaxf(v, __shfl_xor(v, 8, 32));
  return v;
}
static __device__ __forceinline__ float rowsum16(float v) {
  v += __shfl_xor(v, 1, 32);
  v += __shfl_xor(v, 2, 32);
  v += __shfl_xor(v, 4, 32);
  v += __shfl_xor(v, 8, 32);
  return v;
}

// ---------------------------------------------------------------------------
// 1) f32 -> bf16 elementwise conversion
// ---------------------------------------------------------------------------
__global__ void cvt_bf16_kernel(const float* __restrict__ src,
                                bf16* __restrict__ dst, int n) {
  int i = blockIdx.x * blockDim.x + threadIdx.x;
  int stride = gridDim.x * blockDim.x;
  for (; i < n; i += stride) dst[i] = (bf16)src[i];
}

// ---------------------------------------------------------------------------
// 2) Projection GEMM: C[r,c] = sum_k A[r,k] * W[c,k]   (A: ROWSxD, W: DxD)
//    Each wave computes a 32(M) x 64(N) tile (B fragments reused across the
//    two M subtiles -> 1.5 B128 loads per WMMA); K-loop in steps of 32.
//    mode 0: dst[b,h,s,dh] = val*scale           ([B,H,S,Dh] bf16)
//    mode 1: dst[b,h,dh,s] = val*scale           (transposed, for V)
// ---------------------------------------------------------------------------
__global__ __launch_bounds__(256) void proj_kernel(
    const bf16* __restrict__ A, const bf16* __restrict__ W,
    bf16* __restrict__ dst, int mode, float scale) {
  const int lane = threadIdx.x & 31;
  const int wave = threadIdx.x >> 5;
  const int wid  = blockIdx.x * 8 + wave;     // 4096 wave tiles
  const int ntile = wid & 31;                 // 32 tiles of N=64
  const int mtile = wid >> 5;                 // 128 tiles of M=32
  const int m0 = mtile * 32, n0 = ntile * 64;
  const int lm = lane & 15;
  const int lk = (lane >> 4) * 16;

  v8f acc[2][4];
#pragma unroll
  for (int u = 0; u < 2; ++u)
#pragma unroll
    for (int j = 0; j < 4; ++j) acc[u][j] = v8f{};

  const bf16* arow0 = A + (size_t)(m0 + lm) * D_MODEL + lk;
  const bf16* arow1 = arow0 + (size_t)16 * D_MODEL;
  const bf16* brow  = W + (size_t)(n0 + lm) * D_MODEL + lk;

  for (int k0 = 0; k0 < D_MODEL; k0 += 32) {
    v16bf a0 = *(const v16bf*)(arow0 + k0);
    v16bf a1 = *(const v16bf*)(arow1 + k0);
#pragma unroll
    for (int j = 0; j < 4; ++j) {
      v16bf b = *(const v16bf*)(brow + (size_t)j * 16 * D_MODEL + k0);
      acc[0][j] = __builtin_amdgcn_wmma_f32_16x16x32_bf16(
          false, a0, false, b, (short)0, acc[0][j], false, false);
      acc[1][j] = __builtin_amdgcn_wmma_f32_16x16x32_bf16(
          false, a1, false, b, (short)0, acc[1][j], false, false);
    }
  }

  const int mhalf = (lane >> 4) * 8;  // C layout: M = i + 8*(lane/16)
  const int nL = lane & 15;
#pragma unroll
  for (int u = 0; u < 2; ++u) {
#pragma unroll
    for (int j = 0; j < 4; ++j) {
#pragma unroll
      for (int i = 0; i < 8; ++i) {
        int r = m0 + u * 16 + mhalf + i;
        int c = n0 + j * 16 + nL;
        int b = r >> 11, s = r & (SEQ - 1);
        int h = c >> 7, dh = c & (HEAD_DIM - 1);
        float v = acc[u][j][i] * scale;
        size_t idx = (mode == 0)
            ? ((((size_t)b * N_HEADS + h) * SEQ + s) * HEAD_DIM + dh)
            : ((((size_t)b * N_HEADS + h) * HEAD_DIM + dh) * SEQ + s);
        dst[idx] = (bf16)v;
      }
    }
  }
}

// ---------------------------------------------------------------------------
// 3) Flash attention: one wave per 16-row query tile, 32-key tiles.
//    Q,K: [B,H,S,Dh] bf16 (Q pre-scaled).  Vt: [B,H,Dh,S] bf16.
//    alibi: [H,S,S] f32.  Output: [B,S,D] bf16.
// ---------------------------------------------------------------------------
__global__ __launch_bounds__(256) void attn_kernel(
    const bf16* __restrict__ Q, const bf16* __restrict__ K,
    const bf16* __restrict__ Vt, const float* __restrict__ alibi,
    bf16* __restrict__ Obf) {
  __shared__ float ldsP[8][16 * 32];  // per-wave 16x32 probability tile
  const int lane = threadIdx.x & 31;
  const int wave = threadIdx.x >> 5;
  const int wid  = blockIdx.x * 8 + wave;  // 4096 tiles = B*H*(S/16)
  const int qt   = wid & 127;
  const int bh   = wid >> 7;               // 0..31
  const int h    = bh & (N_HEADS - 1);
  const int b    = bh >> 4;
  const int m0   = qt * 16;
  const int lm   = lane & 15;
  const int lk   = (lane >> 4) * 16;
  const int mhalf = (lane >> 4) * 8;
  const int nL    = lane & 15;

  // preload the whole Q tile: 16 rows x 128, 4 chunks of 32 along Dh
  v16bf qa[4];
  const bf16* qbase = Q + ((size_t)bh * SEQ + m0 + lm) * HEAD_DIM + lk;
#pragma unroll
  for (int kk = 0; kk < 4; ++kk) qa[kk] = *(const v16bf*)(qbase + kk * 32);

  v8f acc[8];
#pragma unroll
  for (int j = 0; j < 8; ++j) acc[j] = v8f{};
  float mrow[8], lrow[8];
#pragma unroll
  for (int i = 0; i < 8; ++i) { mrow[i] = -3.0e38f; lrow[i] = 0.0f; }

  const float* arow = alibi + (size_t)h * SEQ * SEQ;
  const int ktmax = (m0 + 15) >> 5;  // inclusive, causal

  for (int kt = 0; kt <= ktmax; ++kt) {
    const int kbase = kt * 32;

    // prefetch next key tile's alibi lines (global_prefetch_b8); the 268MB
    // bias tensor is the dominant single-use HBM stream in this problem.
    if (kt < ktmax) {
      const float* pf = arow + (size_t)(m0 + mhalf) * SEQ + kbase + 32 + nL;
      __builtin_prefetch(pf, 0, 1);
      __builtin_prefetch(pf + (size_t)2 * SEQ, 0, 1);
      __builtin_prefetch(pf + (size_t)4 * SEQ, 0, 1);
      __builtin_prefetch(pf + (size_t)6 * SEQ, 0, 1);
    }

    // ---- scores: S = Q * K^T, two 16x16 N-subtiles, K-dim = 128 ----
    v8f s0 = v8f{}, s1 = v8f{};
    const bf16* kb = K + ((size_t)bh * SEQ + kbase + lm) * HEAD_DIM + lk;
#pragma unroll
    for (int kk = 0; kk < 4; ++kk) {
      v16bf b0 = *(const v16bf*)(kb + kk * 32);
      v16bf b1 = *(const v16bf*)(kb + (size_t)16 * HEAD_DIM + kk * 32);
      s0 = __builtin_amdgcn_wmma_f32_16x16x32_bf16(
          false, qa[kk], false, b0, (short)0, s0, false, false);
      s1 = __builtin_amdgcn_wmma_f32_16x16x32_bf16(
          false, qa[kk], false, b1, (short)0, s1, false, false);
    }

    // ---- bias + causal mask + online softmax update ----
    float mnew[8];
#pragma unroll
    for (int i = 0; i < 8; ++i) {
      int m  = mhalf + i;
      int qg = m0 + m;
      int k0g = kbase + nL;
      int k1g = kbase + 16 + nL;
      const float* ar = arow + (size_t)qg * SEQ;
      float x0 = (k0g <= qg) ? s0[i] + ar[k0g] : -3.0e38f;
      float x1 = (k1g <= qg) ? s1[i] + ar[k1g] : -3.0e38f;
      s0[i] = x0; s1[i] = x1;
      float mx = rowmax16(fmaxf(x0, x1));
      mnew[i] = fmaxf(mrow[i], mx);
    }
#pragma unroll
    for (int i = 0; i < 8; ++i) {
      float p0 = __expf(s0[i] - mnew[i]);
      float p1 = __expf(s1[i] - mnew[i]);
      float sc = __expf(mrow[i] - mnew[i]);
      float rs = rowsum16(p0 + p1);
      lrow[i] = lrow[i] * sc + rs;
      mrow[i] = mnew[i];
#pragma unroll
      for (int j = 0; j < 8; ++j) acc[j][i] *= sc;
      int m = mhalf + i;
      ldsP[wave][m * 32 + nL]      = p0;
      ldsP[wave][m * 32 + 16 + nL] = p1;
    }

    // ---- restage P from C-layout to A-layout via LDS (wave-private,
    //      DS ops are in-order within a wave) ----
    const float* prow = &ldsP[wave][lm * 32 + lk];
    v16bf pa;
#pragma unroll
    for (int e = 0; e < 16; ++e) pa[e] = (bf16)prow[e];

    // ---- O += P * V  (B from Vt: column dh contiguous along keys) ----
    const bf16* vb = Vt + ((size_t)bh * HEAD_DIM + nL) * SEQ + kbase + lk;
#pragma unroll
    for (int j = 0; j < 8; ++j) {
      v16bf bv = *(const v16bf*)(vb + (size_t)j * 16 * SEQ);
      acc[j] = __builtin_amdgcn_wmma_f32_16x16x32_bf16(
          false, pa, false, bv, (short)0, acc[j], false, false);
    }
  }

  // ---- epilogue: normalize and write [B,S,D] bf16 ----
#pragma unroll
  for (int i = 0; i < 8; ++i) {
    float inv = 1.0f / lrow[i];
    int m = mhalf + i;
    size_t rowoff = ((size_t)(b * SEQ + m0 + m)) * D_MODEL + h * HEAD_DIM;
#pragma unroll
    for (int j = 0; j < 8; ++j) {
      Obf[rowoff + j * 16 + nL] = (bf16)(acc[j][i] * inv);
    }
  }
}

// ---------------------------------------------------------------------------
// 4) Output projection GEMM -> f32: out[r,c] = sum_k A[r,k] * Wo[c,k]
//    Same 32x64 wave tiling as proj_kernel.
// ---------------------------------------------------------------------------
__global__ __launch_bounds__(256) void oproj_kernel(
    const bf16* __restrict__ A, const bf16* __restrict__ W,
    float* __restrict__ out) {
  const int lane = threadIdx.x & 31;
  const int wave = threadIdx.x >> 5;
  const int wid  = blockIdx.x * 8 + wave;     // 4096 wave tiles
  const int ntile = wid & 31;
  const int mtile = wid >> 5;
  const int m0 = mtile * 32, n0 = ntile * 64;
  const int lm = lane & 15;
  const int lk = (lane >> 4) * 16;

  v8f acc[2][4];
#pragma unroll
  for (int u = 0; u < 2; ++u)
#pragma unroll
    for (int j = 0; j < 4; ++j) acc[u][j] = v8f{};

  const bf16* arow0 = A + (size_t)(m0 + lm) * D_MODEL + lk;
  const bf16* arow1 = arow0 + (size_t)16 * D_MODEL;
  const bf16* brow  = W + (size_t)(n0 + lm) * D_MODEL + lk;

  for (int k0 = 0; k0 < D_MODEL; k0 += 32) {
    v16bf a0 = *(const v16bf*)(arow0 + k0);
    v16bf a1 = *(const v16bf*)(arow1 + k0);
#pragma unroll
    for (int j = 0; j < 4; ++j) {
      v16bf b = *(const v16bf*)(brow + (size_t)j * 16 * D_MODEL + k0);
      acc[0][j] = __builtin_amdgcn_wmma_f32_16x16x32_bf16(
          false, a0, false, b, (short)0, acc[0][j], false, false);
      acc[1][j] = __builtin_amdgcn_wmma_f32_16x16x32_bf16(
          false, a1, false, b, (short)0, acc[1][j], false, false);
    }
  }

  const int mhalf = (lane >> 4) * 8;
  const int nL = lane & 15;
#pragma unroll
  for (int u = 0; u < 2; ++u) {
#pragma unroll
    for (int j = 0; j < 4; ++j) {
#pragma unroll
      for (int i = 0; i < 8; ++i) {
        int r = m0 + u * 16 + mhalf + i;
        int c = n0 + j * 16 + nL;
        out[(size_t)r * D_MODEL + c] = acc[u][j][i];
      }
    }
  }
}

// ---------------------------------------------------------------------------
// launcher
// ---------------------------------------------------------------------------
extern "C" void kernel_launch(void* const* d_in, const int* in_sizes, int n_in,
                              void* d_out, int out_size, void* d_ws, size_t ws_size,
                              hipStream_t stream) {
  const float* x     = (const float*)d_in[0];
  const float* alibi = (const float*)d_in[1];
  const float* Wq    = (const float*)d_in[2];
  const float* Wk    = (const float*)d_in[3];
  const float* Wv    = (const float*)d_in[4];
  const float* Wo    = (const float*)d_in[5];
  float* out = (float*)d_out;

  // workspace layout (bf16 elements); total ~117 MB
  bf16* Xbf = (bf16*)d_ws;
  bf16* Wqb = Xbf + (size_t)ROWS * D_MODEL;           // 8,388,608
  bf16* Wkb = Wqb + (size_t)D_MODEL * D_MODEL;        // 4,194,304 each
  bf16* Wvb = Wkb + (size_t)D_MODEL * D_MODEL;
  bf16* Wob = Wvb + (size_t)D_MODEL * D_MODEL;
  bf16* Qw  = Wob + (size_t)D_MODEL * D_MODEL;        // [B,H,S,Dh]
  bf16* Kw  = Qw + (size_t)ROWS * D_MODEL;
  bf16* Vt  = Kw + (size_t)ROWS * D_MODEL;            // [B,H,Dh,S]
  bf16* Abf = Vt + (size_t)ROWS * D_MODEL;            // [B,S,D]

  const float qscale = 0.08838834764831845f;          // 1/sqrt(128)

  // 1) conversions
  cvt_bf16_kernel<<<4096, 256, 0, stream>>>(x,  Xbf, ROWS * D_MODEL);
  cvt_bf16_kernel<<<2048, 256, 0, stream>>>(Wq, Wqb, D_MODEL * D_MODEL);
  cvt_bf16_kernel<<<2048, 256, 0, stream>>>(Wk, Wkb, D_MODEL * D_MODEL);
  cvt_bf16_kernel<<<2048, 256, 0, stream>>>(Wv, Wvb, D_MODEL * D_MODEL);
  cvt_bf16_kernel<<<2048, 256, 0, stream>>>(Wo, Wob, D_MODEL * D_MODEL);

  // 2) projections (4096 wave tiles each -> 512 blocks of 8 waves)
  proj_kernel<<<512, 256, 0, stream>>>(Xbf, Wqb, Qw, 0, qscale);
  proj_kernel<<<512, 256, 0, stream>>>(Xbf, Wkb, Kw, 0, 1.0f);
  proj_kernel<<<512, 256, 0, stream>>>(Xbf, Wvb, Vt, 1, 1.0f);

  // 3) flash attention (4096 query tiles -> 512 blocks of 8 waves)
  attn_kernel<<<512, 256, 0, stream>>>(Qw, Kw, Vt, alibi, Abf);

  // 4) output projection -> f32
  oproj_kernel<<<512, 256, 0, stream>>>(Abf, Wob, out);
}
